// ColorGradientNet_52802327937747
// MI455X (gfx1250) — matrix operand order repaired and checked by reading, not compiled, not run on hardware
//
#include <hip/hip_runtime.h>
#include <hip/hip_bf16.h>

typedef __attribute__((ext_vector_type(16))) _Float16 v16h;
typedef __attribute__((ext_vector_type(8)))  _Float16 v8h;
typedef __attribute__((ext_vector_type(2)))  _Float16 v2h;
typedef __attribute__((ext_vector_type(8)))  float    v8f;

#define KNN 20
#define RSQ 0.99999500003749f   // 1/sqrt(1+1e-5)

static __device__ __forceinline__ float lrelu(float v) {
    return v >= 0.0f ? v : 0.2f * v;
}

// Combine the two contiguous 8-half runs of the 16-bit WMMA A layout:
// group g needs K in [8g, 8g+8) then [16+8g, 16+8g+8).
static __device__ __forceinline__ v16h make_a(v8h lo, v8h hi) {
    return __builtin_shufflevector(lo, hi, 0, 1, 2, 3, 4, 5, 6, 7,
                                           8, 9, 10, 11, 12, 13, 14, 15);
}

// ---------------------------------------------------------------------------
// Kernel 1: pairwise-distance top-20 (pd = 2*inner - |xi|^2 - |xj|^2).
// One thread per point; xyz + squared norms staged in LDS (64KB).
// ---------------------------------------------------------------------------
__global__ __launch_bounds__(256) void knn_kernel(
    const float* __restrict__ x, int* __restrict__ idx, int Nn)
{
    __shared__ float sx[4096], sy[4096], sz[4096], sq[4096];
    const int b = blockIdx.y;
    const float* xb = x + (size_t)b * 10 * Nn;
    for (int i = threadIdx.x; i < Nn; i += 256) {
        float a0 = xb[i], a1 = xb[Nn + i], a2 = xb[2 * Nn + i];
        sx[i] = a0; sy[i] = a1; sz[i] = a2;
        sq[i] = a0 * a0 + a1 * a1 + a2 * a2;
    }
    __syncthreads();

    const int n = blockIdx.x * 256 + threadIdx.x;
    const float px = sx[n], py = sy[n], pz = sz[n], pq = sq[n];

    float best[KNN];
    int   bidx[KNN];
#pragma unroll
    for (int i = 0; i < KNN; ++i) { best[i] = -3.4e38f; bidx[i] = 0; }

    for (int m = 0; m < Nn; ++m) {
        float d = 2.0f * (px * sx[m] + py * sy[m] + pz * sz[m]) - pq - sq[m];
        if (d > best[KNN - 1]) {
            int pos = KNN - 1;
#pragma unroll
            for (int j = KNN - 2; j >= 0; --j) {
                if (d > best[j]) {
                    best[j + 1] = best[j];
                    bidx[j + 1] = bidx[j];
                    pos = j;
                }
            }
#pragma unroll
            for (int j = 0; j < KNN; ++j)
                if (j == pos) { best[j] = d; bidx[j] = m; }
        }
    }
    int* op = idx + ((size_t)b * Nn + n) * KNN;
#pragma unroll
    for (int i = 0; i < KNN; ++i) op[i] = bidx[i];
}

// ---------------------------------------------------------------------------
// Kernel 2: fused edgeconv (9->64 -> 64->64 -> max over k). One wave per
// point; 32 rows (20 neighbors + 12 copies of neighbor 0, max-safe).
// Feature tile built in LDS with compile-time channel indices; fragments
// loaded as contiguous b128 runs per the ISA 16-bit A/B layouts.
// ---------------------------------------------------------------------------
__global__ __launch_bounds__(128) void edgeconv_kernel(
    const float* __restrict__ x, const int* __restrict__ idx,
    const float* __restrict__ w1, const float* __restrict__ g1, const float* __restrict__ b1,
    const float* __restrict__ w2, const float* __restrict__ g2, const float* __restrict__ b2,
    float* __restrict__ x1out, int Nn)
{
    __shared__ _Float16 w1s[64][32];     // [O][K pad 9->32]
    __shared__ _Float16 w2s[64][64];     // [O][K]
    __shared__ _Float16 fs[4][32][32];   // per-wave feature tile [row][K]
    __shared__ _Float16 hs[4][32][64];   // per-wave h1 tile [row][O]

    const int tid = threadIdx.x;
    const int lane = tid & 31, wave = tid >> 5;
    const int g = lane >> 4, mr = lane & 15;

    for (int i = tid; i < 64 * 16; i += 128) {
        int o = i >> 4, kk2 = (i & 15) * 2;
        v2h pk;
        pk[0] = (_Float16)((kk2     < 9) ? w1[o * 9 + kk2]     : 0.0f);
        pk[1] = (_Float16)((kk2 + 1 < 9) ? w1[o * 9 + kk2 + 1] : 0.0f);
        *(v2h*)&w1s[o][kk2] = pk;
    }
    for (int i = tid; i < 64 * 32; i += 128) {
        int o = i >> 5, kk2 = (i & 31) * 2;
        v2h pk;
        pk[0] = (_Float16)w2[o * 64 + kk2];
        pk[1] = (_Float16)w2[o * 64 + kk2 + 1];
        *(v2h*)&w2s[o][kk2] = pk;
    }

    const int p = blockIdx.x * 4 + wave;
    const int b = p / Nn, n = p % Nn;
    const float* xb = x + (size_t)b * 10 * Nn;
    const int* idp = idx + (size_t)p * KNN;

    // Build feature row r = lane: [xyz(3) | rgb(3) | rgb - nbr_rgb(3) | 0...]
    {
        int j = (lane < KNN) ? lane : 0;
        int nb = idp[j];
        _Float16 fv[32];
#pragma unroll
        for (int kk = 0; kk < 32; ++kk) fv[kk] = (_Float16)0.0f;
#pragma unroll
        for (int c = 0; c < 3; ++c) {
            float ctr = xb[c * Nn + n];
            float rgb = xb[(3 + c) * Nn + n];
            float nr  = xb[(3 + c) * Nn + nb];
            fv[c]     = (_Float16)ctr;
            fv[3 + c] = (_Float16)rgb;
            fv[6 + c] = (_Float16)(rgb - nr);
        }
#pragma unroll
        for (int q = 0; q < 16; ++q) {
            v2h pk; pk[0] = fv[2 * q]; pk[1] = fv[2 * q + 1];
            *(v2h*)&fs[wave][lane][2 * q] = pk;
        }
    }
    __syncthreads();

    // --- GEMM1: f(32x32) @ w1^T -> 32x64 ---
    v8f acc1[2][4] = {};
    v16h bf1[4];
#pragma unroll
    for (int t = 0; t < 4; ++t)
        bf1[t] = make_a(*(const v8h*)&w1s[t * 16 + mr][8 * g], *(const v8h*)&w1s[t * 16 + mr][16 + 8 * g]);
#pragma unroll
    for (int mt = 0; mt < 2; ++mt) {
        int row = mt * 16 + mr;
        v8h lo = *(const v8h*)&fs[wave][row][8 * g];
        v8h hi = *(const v8h*)&fs[wave][row][16 + 8 * g];
        v16h a = make_a(lo, hi);
#pragma unroll
        for (int t = 0; t < 4; ++t)
            acc1[mt][t] = __builtin_amdgcn_wmma_f32_16x16x32_f16(
                false, a, false, bf1[t], (short)0, acc1[mt][t], false, false);
    }

    // BN1 + lrelu, write h1 tile to LDS as f16 (column writes, scalar)
#pragma unroll
    for (int t = 0; t < 4; ++t) {
        int o = t * 16 + mr;
        float s = g1[o] * RSQ, sh = b1[o];
#pragma unroll
        for (int mt = 0; mt < 2; ++mt)
#pragma unroll
            for (int r = 0; r < 8; ++r) {
                float v = lrelu(acc1[mt][t][r] * s + sh);
                hs[wave][mt * 16 + r + 8 * g][o] = (_Float16)v;
            }
    }
    __syncthreads();

    // --- GEMM2: h1(32x64) @ w2^T -> 32x64 ---
    v8f acc2[2][4] = {};
#pragma unroll
    for (int kc = 0; kc < 2; ++kc) {
        v16h a2[2];
#pragma unroll
        for (int mt = 0; mt < 2; ++mt) {
            int row = mt * 16 + mr;
            v8h lo = *(const v8h*)&hs[wave][row][kc * 32 + 8 * g];
            v8h hi = *(const v8h*)&hs[wave][row][kc * 32 + 16 + 8 * g];
            a2[mt] = make_a(lo, hi);
        }
#pragma unroll
        for (int t = 0; t < 4; ++t) {
            v16h bf = make_a(*(const v8h*)&w2s[t * 16 + mr][kc * 32 + 8 * g], *(const v8h*)&w2s[t * 16 + mr][kc * 32 + 16 + 8 * g]);
#pragma unroll
            for (int mt = 0; mt < 2; ++mt)
                acc2[mt][t] = __builtin_amdgcn_wmma_f32_16x16x32_f16(
                    false, a2[mt], false, bf, (short)0, acc2[mt][t], false, false);
        }
    }

    // BN2 + lrelu + column max over 32 rows -> x1[b][n][64]
#pragma unroll
    for (int t = 0; t < 4; ++t) {
        int o = t * 16 + mr;
        float s = g2[o] * RSQ, sh = b2[o];
        float m = -3.4e38f;
#pragma unroll
        for (int mt = 0; mt < 2; ++mt)
#pragma unroll
            for (int r = 0; r < 8; ++r)
                m = fmaxf(m, lrelu(acc2[mt][t][r] * s + sh));
        m = fmaxf(m, __shfl_xor(m, 16, 32));
        if (g == 0) x1out[(size_t)p * 64 + o] = m;
    }
}

// ---------------------------------------------------------------------------
// Kernel 3: generic WMMA GEMM: out = act(bn(A @ W^T + linb)).
// Block = 8 waves -> 128 rows x 64 cols, K streamed in 32-chunks via LDS.
// MODE 0: A = [x1,x1] duplicated (K=128); MODE 1: A plain (R x K);
// MODE 2: A = concat[xg(1024)|x3(256)|x2(128)|x1(64)] gathered on the fly.
// ---------------------------------------------------------------------------
template <int MODE>
__global__ __launch_bounds__(256) void gemm_kernel(
    const float* __restrict__ A0, const float* __restrict__ A1,
    const float* __restrict__ A2, const float* __restrict__ A3,
    const float* __restrict__ W,  const float* __restrict__ linb,
    const float* __restrict__ gbn, const float* __restrict__ bbn,
    float* __restrict__ out, int Kdim, int OUT, int applyLrelu)
{
    __shared__ _Float16 As[128][32];   // [row][K]
    __shared__ _Float16 Ws[64][32];    // [O][K]  (K-contiguous for b128 frags)

    const int tid = threadIdx.x;
    const int lane = tid & 31, wave = tid >> 5;
    const int g = lane >> 4, mr = lane & 15;
    const int rowBase = blockIdx.x * 128;
    const int colBase = blockIdx.y * 64;

    v8f acc[4] = {};
    const int nch = Kdim >> 5;

    for (int kc = 0; kc < nch; ++kc) {
        const int k0 = kc * 32;
        // Stage A chunk: 128x32, two halves per thread-step (v2h packed)
        for (int i = tid; i < 128 * 16; i += 256) {
            int rl = i >> 4, kk2 = (i & 15) * 2;
            int row = rowBase + rl;
            int c = k0 + kk2;
            const float* src;
            if (MODE == 0) {
                src = &A0[(size_t)row * 64 + (c & 63)];
            } else if (MODE == 1) {
                src = &A0[(size_t)row * Kdim + c];
            } else {
                if (c < 1024)      src = &A0[(size_t)(row >> 12) * 1024 + c]; // xg, N==4096
                else if (c < 1280) src = &A1[(size_t)row * 256 + (c - 1024)];
                else if (c < 1408) src = &A2[(size_t)row * 128 + (c - 1280)];
                else               src = &A3[(size_t)row * 64  + (c - 1408)];
            }
            v2h pk; pk[0] = (_Float16)src[0]; pk[1] = (_Float16)src[1];
            *(v2h*)&As[rl][kk2] = pk;
        }
        // Stage W chunk: 64x32
        for (int i = tid; i < 64 * 16; i += 256) {
            int o = i >> 4, kk2 = (i & 15) * 2;
            const float* wp = &W[(size_t)(colBase + o) * Kdim + k0 + kk2];
            v2h pk; pk[0] = (_Float16)wp[0]; pk[1] = (_Float16)wp[1];
            *(v2h*)&Ws[o][kk2] = pk;
        }
        __syncthreads();

        int rowL = wave * 16 + mr;
        v8h lo = *(const v8h*)&As[rowL][8 * g];
        v8h hi = *(const v8h*)&As[rowL][16 + 8 * g];
        v16h a = make_a(lo, hi);
#pragma unroll
        for (int t = 0; t < 4; ++t) {
            v16h bf = make_a(*(const v8h*)&Ws[t * 16 + mr][8 * g], *(const v8h*)&Ws[t * 16 + mr][16 + 8 * g]);
            acc[t] = __builtin_amdgcn_wmma_f32_16x16x32_f16(
                false, a, false, bf, (short)0, acc[t], false, false);
        }
        __syncthreads();
    }

#pragma unroll
    for (int t = 0; t < 4; ++t) {
        int o = colBase + t * 16 + mr;
        float add = linb ? linb[o] : 0.0f;
        float s = gbn ? gbn[o] * RSQ : 1.0f;
        float sh = bbn ? bbn[o] : 0.0f;
#pragma unroll
        for (int r = 0; r < 8; ++r) {
            int row = rowBase + wave * 16 + r + 8 * g;
            float v = (acc[t][r] + add) * s + sh;
            if (applyLrelu) v = lrelu(v);
            out[(size_t)row * OUT + o] = v;
        }
    }
}

// ---------------------------------------------------------------------------
// Small VALU kernels: global max over N, w5 layer, final 256->2 + transpose.
// ---------------------------------------------------------------------------
__global__ __launch_bounds__(256) void gmax_kernel(
    const float* __restrict__ x3, float* __restrict__ x4, int Nn)
{
    int b = blockIdx.x, o = threadIdx.x;
    const float* p = x3 + (size_t)b * Nn * 256 + o;
    float m = -3.4e38f;
    for (int n = 0; n < Nn; ++n) m = fmaxf(m, p[(size_t)n * 256]);
    x4[b * 256 + o] = m;
}

__global__ __launch_bounds__(256) void xg_kernel(
    const float* __restrict__ x4, const float* __restrict__ w5,
    const float* __restrict__ g5, const float* __restrict__ b5,
    float* __restrict__ xg)
{
    int i = blockIdx.x * 256 + threadIdx.x;   // B*1024
    int b = i >> 10, o = i & 1023;
    float s = 0.0f;
    for (int c = 0; c < 256; ++c) s += x4[b * 256 + c] * w5[o * 256 + c];
    xg[i] = lrelu(s * (g5[o] * RSQ) + b5[o]);
}

__global__ __launch_bounds__(256) void final_kernel(
    const float* __restrict__ h2, const float* __restrict__ l3w,
    const float* __restrict__ l3b, float* __restrict__ out, int Nn)
{
    int p = blockIdx.x * 256 + threadIdx.x;   // B*N
    int b = p / Nn, n = p % Nn;
    const float* hp = h2 + (size_t)p * 256;
#pragma unroll
    for (int o = 0; o < 2; ++o) {
        float s = l3b[o];
        for (int c = 0; c < 256; ++c) s += hp[c] * l3w[o * 256 + c];
        out[((size_t)b * 2 + o) * Nn + n] = s;
    }
}

// ---------------------------------------------------------------------------
extern "C" void kernel_launch(void* const* d_in, const int* in_sizes, int n_in,
                              void* d_out, int out_size, void* d_ws, size_t ws_size,
                              hipStream_t stream) {
    (void)in_sizes; (void)n_in; (void)out_size; (void)ws_size;
    const float* x   = (const float*)d_in[0];
    // d_in[1] = k (int scalar) == 20, compile-time constant here
    const float* w1  = (const float*)d_in[2];
    const float* g1  = (const float*)d_in[3];
    const float* b1  = (const float*)d_in[4];
    const float* w2  = (const float*)d_in[5];
    const float* g2  = (const float*)d_in[6];
    const float* b2  = (const float*)d_in[7];
    const float* w3  = (const float*)d_in[8];
    const float* g3  = (const float*)d_in[9];
    const float* b3  = (const float*)d_in[10];
    const float* w4  = (const float*)d_in[11];
    const float* g4  = (const float*)d_in[12];
    const float* b4  = (const float*)d_in[13];
    const float* w5  = (const float*)d_in[14];
    const float* g5  = (const float*)d_in[15];
    const float* b5  = (const float*)d_in[16];
    const float* l1w = (const float*)d_in[17];
    const float* l1b = (const float*)d_in[18];
    const float* g6  = (const float*)d_in[19];
    const float* b6  = (const float*)d_in[20];
    const float* l2w = (const float*)d_in[21];
    const float* l2b = (const float*)d_in[22];
    const float* g7  = (const float*)d_in[23];
    const float* b7  = (const float*)d_in[24];
    const float* l3w = (const float*)d_in[25];
    const float* l3b = (const float*)d_in[26];

    const int B = 8, N = 4096;
    char* wsb = (char*)d_ws;
    size_t off = 0;
    auto alloc = [&](size_t bytes) -> void* {
        void* p = wsb + off;
        off = (off + bytes + 255) & ~(size_t)255;
        return p;
    };
    int*   idx = (int*)  alloc((size_t)B * N * KNN * 4);
    float* x1  = (float*)alloc((size_t)B * N * 64  * 4);
    float* x2  = (float*)alloc((size_t)B * N * 128 * 4);
    float* x3  = (float*)alloc((size_t)B * N * 256 * 4);
    float* x4  = (float*)alloc((size_t)B * 256 * 4);
    float* xg  = (float*)alloc((size_t)B * 1024 * 4);
    float* h1  = (float*)alloc((size_t)B * N * 512 * 4);
    float* h2  = (float*)alloc((size_t)B * N * 256 * 4);

    const int R = B * N;

    knn_kernel<<<dim3(N / 256, B), 256, 0, stream>>>(x, idx, N);
    edgeconv_kernel<<<R / 4, 128, 0, stream>>>(x, idx, w1, g1, b1, w2, g2, b2, x1, N);
    gemm_kernel<0><<<dim3(R / 128, 128 / 64), 256, 0, stream>>>(
        x1, nullptr, nullptr, nullptr, w3, nullptr, g3, b3, x2, 128, 128, 1);
    gemm_kernel<1><<<dim3(R / 128, 256 / 64), 256, 0, stream>>>(
        x2, nullptr, nullptr, nullptr, w4, nullptr, g4, b4, x3, 128, 256, 1);
    gmax_kernel<<<B, 256, 0, stream>>>(x3, x4, N);
    xg_kernel<<<B * 1024 / 256, 256, 0, stream>>>(x4, w5, g5, b5, xg);
    gemm_kernel<2><<<dim3(R / 128, 512 / 64), 256, 0, stream>>>(
        xg, x3, x2, x1, l1w, l1b, g6, b6, h1, 1472, 512, 1);
    gemm_kernel<1><<<dim3(R / 128, 256 / 64), 256, 0, stream>>>(
        h1, nullptr, nullptr, nullptr, l2w, l2b, g7, b7, h2, 512, 256, 1);
    final_kernel<<<R / 256, 256, 0, stream>>>(h2, l3w, l3b, (float*)d_out, N);
}
